// EmotionGNN_4827543241130
// MI455X (gfx1250) — matrix operand order repaired and checked by reading, not compiled, not run on hardware
//
#include <hip/hip_runtime.h>
#include <hip/hip_bf16.h>

// ---------------------------------------------------------------------------
// Types for CDNA5 WMMA (wave32): v_wmma_f32_16x16x32_bf16
// ---------------------------------------------------------------------------
typedef __attribute__((ext_vector_type(16))) __bf16 v16bf;
typedef __attribute__((ext_vector_type(8)))  float  v8f;

__device__ __forceinline__ __bf16 f2bf(float f) {
    unsigned u = __builtin_bit_cast(unsigned, f);
    u += 0x7FFFu + ((u >> 16) & 1u);          // round-to-nearest-even
    unsigned short h = (unsigned short)(u >> 16);
    return __builtin_bit_cast(__bf16, h);
}

// ---------------------------------------------------------------------------
// Small utility kernels
// ---------------------------------------------------------------------------
__global__ void init_f32_kernel(float* __restrict__ p, float v, long long n) {
    long long i = (long long)blockIdx.x * blockDim.x + threadIdx.x;
    if (i < n) p[i] = v;
}

__global__ void degree_kernel(const long long* __restrict__ dst,
                              float* __restrict__ deg, int E) {
    int e = blockIdx.x * blockDim.x + threadIdx.x;
    if (e < E) atomicAdd(&deg[dst[e]], 1.0f);
}

__global__ void dinv_kernel(const float* __restrict__ deg,
                            float* __restrict__ dinv, int n) {
    int i = blockIdx.x * blockDim.x + threadIdx.x;
    if (i < n) dinv[i] = rsqrtf(deg[i]);      // deg >= 1 always (self-loops)
}

// ---------------------------------------------------------------------------
// Fragment packing helpers
//
// A-fragment (16x32 bf16, MxK): lane = half*16 + mlane, element j.
//   kk(K offset within 32-chunk) = (j&7) + half*8 + (j>>3)*16
//   packed idx = ((mt*(K/32)+kc)*32 + lane)*16 + j
//
// B-fragment (32x16 bf16, KxN): lane l holds N = l&15, K = (j&7) + (l>>4)*8 + (j>>3)*16.
//   packed idx = ((kc*NTILES + nt)*32 + lane)*16 + j
// ---------------------------------------------------------------------------

// Pack row-major f32 weight [K,N] into bf16 WMMA B-fragment order.
__global__ void pack_weights_kernel(const float* __restrict__ W,
                                    __bf16* __restrict__ Bp, int K, int N) {
    int idx = blockIdx.x * blockDim.x + threadIdx.x;
    if (idx >= K * N) return;
    int j    = idx & 15;
    int lane = (idx >> 4) & 31;
    int rest = idx >> 9;
    int ntiles = N >> 4;
    int nt = rest % ntiles;
    int kc = rest / ntiles;
    int n = nt * 16 + (lane & 15);
    int k = kc * 32 + (j & 7) + (lane >> 4) * 8 + (j >> 3) * 16;
    Bp[idx] = f2bf(W[(size_t)k * N + n]);
}

// Pack row-major f32 activations [M,K] into bf16 WMMA A-fragment order.
template <int K>
__global__ void pack_A_kernel(const float* __restrict__ A,
                              __bf16* __restrict__ Ap, int M) {
    long long idx = (long long)blockIdx.x * blockDim.x + threadIdx.x;
    if (idx >= (long long)M * K) return;
    int j    = (int)(idx & 15);
    int lane = (int)((idx >> 4) & 31);
    long long rest = idx >> 9;                 // mt*(K/32) + kc
    int kc = (int)(rest % (K / 32));
    long long mt = rest / (K / 32);
    int half = lane >> 4, mlane = lane & 15;
    int kk = (j & 7) + half * 8 + (j >> 3) * 16;
    long long row = mt * 16 + mlane;
    int k = kc * 32 + kk;
    Ap[idx] = f2bf(A[row * K + k]);
}

// ---------------------------------------------------------------------------
// WMMA GEMM with LDS-staged B panel.
//
// Block = 8 waves, all sharing one N-group (NTW 16-wide tiles), covering
// 8 consecutive M-tiles. The block's entire B panel (KC*NTW fragments,
// 1 KB each) is cooperatively staged in LDS once, then each wave streams
// its A fragments from global (prefetched) and B from low-latency LDS.
// ---------------------------------------------------------------------------
template <int K, int N, int NTW, bool BIAS>
__global__ __launch_bounds__(256)
void gemm_wmma_kernel(const __bf16* __restrict__ Ap,
                      const __bf16* __restrict__ Bp,
                      const float* __restrict__ bias,
                      float* __restrict__ C, int M) {
    constexpr int NTILES  = N / 16;
    constexpr int NGROUPS = NTILES / NTW;
    constexpr int KC      = K / 32;
    constexpr int NFRAG   = KC * NTW;          // fragments staged per block
    __shared__ __bf16 Bs[NFRAG * 512];         // 512 bf16 (1 KB) per fragment

    int wid  = threadIdx.x >> 5;               // 0..7
    int lane = threadIdx.x & 31;
    int mb = blockIdx.x / NGROUPS;
    int ng = blockIdx.x % NGROUPS;

    int mt = mb * 8 + wid;
    bool active = mt < (M / 16);
    if (!active) mt = (M / 16) - 1;            // clamp for safe addressing

    // --- cooperative B panel load: global(b128) -> LDS(b128) -------------
    {
        uint4* ldst = (uint4*)Bs;
        for (int c = threadIdx.x; c < NFRAG * 64; c += 256) {   // 16B chunks
            int frag   = c >> 6;               // kc*NTW + t
            int within = c & 63;
            int kc = frag / NTW, t = frag % NTW;
            const uint4* srcf = (const uint4*)(Bp +
                ((size_t)kc * NTILES + (size_t)ng * NTW + t) * 512);
            ldst[c] = srcf[within];
        }
    }
    __syncthreads();

    // --- main loop: A prefetched from global, B served from LDS ----------
    const v16bf* Aw = (const v16bf*)Ap + (size_t)mt * KC * 32 + lane;
    const v16bf* Bl = (const v16bf*)Bs + lane;

    v8f acc[NTW] = {};
    v16bf a_cur = Aw[0];

#pragma unroll
    for (int kc = 0; kc < KC; ++kc) {
        v16bf a_nxt;
        if (kc + 1 < KC) a_nxt = Aw[(size_t)(kc + 1) * 32];
#pragma unroll
        for (int t = 0; t < NTW; ++t) {
            v16bf bf = Bl[((size_t)kc * NTW + t) * 32];
            acc[t] = __builtin_amdgcn_wmma_f32_16x16x32_bf16(
                false, a_cur, false, bf, (short)0, acc[t], false, false);
        }
        if (kc + 1 < KC) a_cur = a_nxt;
    }

    if (!active) return;

    // C/D layout: VGPR r, lanes 0-15 -> M=r, lanes 16-31 -> M=r+8; N = lane&15
    int kHalf = lane >> 4;
    int mlane = lane & 15;
    int rowBase = mt * 16 + kHalf * 8;
#pragma unroll
    for (int t = 0; t < NTW; ++t) {
        int col = (ng * NTW + t) * 16 + mlane;
        float ba = 0.0f;
        if constexpr (BIAS) ba = bias[col];
#pragma unroll
        for (int r = 0; r < 8; ++r) {
            float v = acc[t][r];
            if constexpr (BIAS) v += ba;
            C[(size_t)(rowBase + r) * N + col] = v;
        }
    }
}

// ---------------------------------------------------------------------------
// Edge scatter: one wave32 per edge; agg[dst] += h[src] * dinv[src]*dinv[dst]
// ---------------------------------------------------------------------------
template <int F>
__global__ __launch_bounds__(256)
void scatter_kernel(const long long* __restrict__ src,
                    const long long* __restrict__ dst,
                    const float* __restrict__ dinv,
                    const float* __restrict__ h,
                    float* __restrict__ agg, int E) {
    int wave = (int)((blockIdx.x * blockDim.x + threadIdx.x) >> 5);
    int lane = threadIdx.x & 31;
    if (wave >= E) return;
    long long s = src[wave];
    long long d = dst[wave];
    float norm = dinv[s] * dinv[d];
    const float4* hp = (const float4*)(h + (size_t)s * F);
    float*        ap = agg + (size_t)d * F;
#pragma unroll
    for (int i = lane; i < F / 4; i += 32) {
        float4 v = hp[i];
        atomicAdd(&ap[4 * i + 0], v.x * norm);
        atomicAdd(&ap[4 * i + 1], v.y * norm);
        atomicAdd(&ap[4 * i + 2], v.z * norm);
        atomicAdd(&ap[4 * i + 3], v.w * norm);
    }
}

// ---------------------------------------------------------------------------
// Fused self-loop + bias (+ReLU) + bf16 A-fragment repack:
// hP(packed) = act(agg[i,f] + h[i,f]*dinv[i]^2 + b[f])
// ---------------------------------------------------------------------------
template <int F, bool RELU>
__global__ void finalize_pack_kernel(const float* __restrict__ h,
                                     const float* __restrict__ dinv,
                                     const float* __restrict__ b,
                                     const float* __restrict__ agg,
                                     __bf16* __restrict__ hP, int n) {
    long long idx = (long long)blockIdx.x * blockDim.x + threadIdx.x;
    if (idx >= (long long)n * F) return;
    int j    = (int)(idx & 15);
    int lane = (int)((idx >> 4) & 31);
    long long rest = idx >> 9;                 // mt*(F/32) + kc
    int kc = (int)(rest % (F / 32));
    long long mt = rest / (F / 32);
    int half = lane >> 4, mlane = lane & 15;
    int kk = (j & 7) + half * 8 + (j >> 3) * 16;
    long long i = mt * 16 + mlane;
    int f = kc * 32 + kk;
    long long lin = i * F + f;
    float di = dinv[i];
    float v = agg[lin] + h[lin] * di * di + b[f];
    if (RELU) v = fmaxf(v, 0.0f);
    hP[idx] = f2bf(v);
}

// ---------------------------------------------------------------------------
// Host-side launch
// ---------------------------------------------------------------------------
static inline size_t alignup(size_t x) { return (x + 255) & ~(size_t)255; }

extern "C" void kernel_launch(void* const* d_in, const int* in_sizes, int n_in,
                              void* d_out, int out_size, void* d_ws, size_t ws_size,
                              hipStream_t stream) {
    (void)n_in; (void)out_size; (void)ws_size;
    constexpr int IN_D = 512, HID_D = 256, OUT_D = 128, PROJ_D = 768;

    const float*     x   = (const float*)d_in[0];
    const long long* ei  = (const long long*)d_in[1];
    const float*     W1  = (const float*)d_in[2];
    const float*     b1  = (const float*)d_in[3];
    const float*     W2  = (const float*)d_in[4];
    const float*     b2  = (const float*)d_in[5];
    const float*     Wp  = (const float*)d_in[6];
    const float*     bp  = (const float*)d_in[7];
    float*           out = (float*)d_out;

    const int n = in_sizes[0] / IN_D;     // 50000
    const int E = in_sizes[1] / 2;        // 800000
    const long long* srcI = ei;
    const long long* dstI = ei + E;

    // Workspace carve-up
    char* ws = (char*)d_ws;
    size_t off = 0;
    float* deg  = (float*)(ws + off); off = alignup(off + (size_t)n * 4);
    float* dinv = (float*)(ws + off); off = alignup(off + (size_t)n * 4);
    __bf16* W1b = (__bf16*)(ws + off); off = alignup(off + (size_t)IN_D  * HID_D  * 2);
    __bf16* W2b = (__bf16*)(ws + off); off = alignup(off + (size_t)HID_D * OUT_D  * 2);
    __bf16* Wpb = (__bf16*)(ws + off); off = alignup(off + (size_t)OUT_D * PROJ_D * 2);
    __bf16* xP  = (__bf16*)(ws + off); off = alignup(off + (size_t)n * IN_D  * 2);
    float* h1   = (float*)(ws + off);  off = alignup(off + (size_t)n * HID_D * 4);
    float* agg1 = (float*)(ws + off);  off = alignup(off + (size_t)n * HID_D * 4);
    __bf16* h1P = (__bf16*)(ws + off); off = alignup(off + (size_t)n * HID_D * 2);
    float* h2   = (float*)(ws + off);  off = alignup(off + (size_t)n * OUT_D * 4);
    float* agg2 = (float*)(ws + off);  off = alignup(off + (size_t)n * OUT_D * 4);
    __bf16* h2P = (__bf16*)(ws + off); off = alignup(off + (size_t)n * OUT_D * 2);

    const int TB = 256;
    auto blocks1d = [](long long work, int tb) { return (unsigned)((work + tb - 1) / tb); };
    const int mtBlocks = (n / 16 + 7) / 8;     // 8 M-tiles per block

    // 1. degrees (self-loop contributes 1 to every node)
    init_f32_kernel<<<blocks1d(n, TB), TB, 0, stream>>>(deg, 1.0f, n);
    degree_kernel<<<blocks1d(E, TB), TB, 0, stream>>>(dstI, deg, E);
    dinv_kernel<<<blocks1d(n, TB), TB, 0, stream>>>(deg, dinv, n);

    // 2. pack weights + layer-1 input to bf16 fragment layouts
    pack_weights_kernel<<<blocks1d((long long)IN_D * HID_D, TB), TB, 0, stream>>>(W1, W1b, IN_D, HID_D);
    pack_weights_kernel<<<blocks1d((long long)HID_D * OUT_D, TB), TB, 0, stream>>>(W2, W2b, HID_D, OUT_D);
    pack_weights_kernel<<<blocks1d((long long)OUT_D * PROJ_D, TB), TB, 0, stream>>>(Wp, Wpb, OUT_D, PROJ_D);
    pack_A_kernel<IN_D><<<blocks1d((long long)n * IN_D, TB), TB, 0, stream>>>(x, xP, n);

    // 3. layer 1: h1 = x @ W1 ; aggregate ; fused bias+ReLU+repack
    gemm_wmma_kernel<IN_D, HID_D, 4, false>
        <<<(unsigned)(mtBlocks * (HID_D / 64)), TB, 0, stream>>>(xP, W1b, nullptr, h1, n);
    init_f32_kernel<<<blocks1d((long long)n * HID_D, TB), TB, 0, stream>>>(agg1, 0.0f, (long long)n * HID_D);
    scatter_kernel<HID_D><<<blocks1d((long long)E * 32, TB), TB, 0, stream>>>(srcI, dstI, dinv, h1, agg1, E);
    finalize_pack_kernel<HID_D, true>
        <<<blocks1d((long long)n * HID_D, TB), TB, 0, stream>>>(h1, dinv, b1, agg1, h1P, n);

    // 4. layer 2: h2 = h1P @ W2 ; aggregate ; fused bias+repack
    gemm_wmma_kernel<HID_D, OUT_D, 4, false>
        <<<(unsigned)(mtBlocks * (OUT_D / 64)), TB, 0, stream>>>(h1P, W2b, nullptr, h2, n);
    init_f32_kernel<<<blocks1d((long long)n * OUT_D, TB), TB, 0, stream>>>(agg2, 0.0f, (long long)n * OUT_D);
    scatter_kernel<OUT_D><<<blocks1d((long long)E * 32, TB), TB, 0, stream>>>(srcI, dstI, dinv, h2, agg2, E);
    finalize_pack_kernel<OUT_D, false>
        <<<blocks1d((long long)n * OUT_D, TB), TB, 0, stream>>>(h2, dinv, b2, agg2, h2P, n);

    // 5. projection: out = h2P @ Wp + bp
    gemm_wmma_kernel<OUT_D, PROJ_D, 4, true>
        <<<(unsigned)(mtBlocks * (PROJ_D / 64)), TB, 0, stream>>>(h2P, Wpb, bp, out, n);
}
